// CriterionMatching_82600811036673
// MI455X (gfx1250) — compile-verified, matrix operand order checked by reading, not run on hardware
//
#include <hip/hip_runtime.h>
#include <hip/hip_bf16.h>

typedef __attribute__((ext_vector_type(8)))  _Float16 half8;
typedef __attribute__((ext_vector_type(16))) _Float16 half16;
typedef __attribute__((ext_vector_type(8)))  float    float8;
typedef unsigned int u32;

// Problem constants (from reference setup_inputs)
constexpr int Bn  = 16;
constexpr int Hh  = 768, Ww = 768;
constexpr int HW  = Hh * Ww;          // 589824
constexpr int Heh = 192, Wew = 192;
constexpr int Dd  = 64;
constexpr int P   = Heh * Wew;        // 36864
constexpr int TOPK = 400;
constexpr int NTILE = TOPK / 16;      // 25
constexpr int IGNORE_IDX = 255;

// Workspace layout (bytes)
constexpr size_t OFF_CESUM = 0;                       // f32[16]
constexpr size_t OFF_CECNT = 64;                      // f32[16]
constexpr size_t OFF_CNT1  = 128;                     // u32[32]
constexpr size_t OFF_CNT2  = 256;                     // u32[32]
constexpr size_t OFF_PAIR  = 384;                     // f32[16]
constexpr size_t OFF_SUM2  = 512;                     // f32[32*64] = 8192B
constexpr size_t OFF_THR   = 8704;                    // u32[32]
constexpr size_t OFF_RNEED = 8832;                    // u32[32]
constexpr size_t ACC_BYTES = 16384;                   // region zeroed each call
constexpr size_t KEY_BYTES = (size_t)32 * P * 4;      // 4,718,592
constexpr size_t OFF_KEYS  = ACC_BYTES;
constexpr size_t OFF_RNORM = OFF_KEYS + KEY_BYTES;
constexpr size_t OFF_M2    = OFF_RNORM + KEY_BYTES;   // u8[32*P]
constexpr size_t OFF_SMAT  = (OFF_M2 + (size_t)32 * P + 255) & ~(size_t)255; // f16[32*400*64]

__device__ __forceinline__ u32 f2sortable(float f) {
  u32 u = __float_as_uint(f);
  return (u & 0x80000000u) ? ~u : (u | 0x80000000u);
}

// ---------------- Cross-entropy: per-sample sum + valid count ----------------
__global__ void ce_kernel(const float* __restrict__ outp, const int* __restrict__ lab,
                          float* ce_sum, float* ce_cnt) {
  int b   = blockIdx.y;
  int pix = blockIdx.x * blockDim.x + threadIdx.x;   // grid.x * 256 == HW exactly
  const float* o = outp + (size_t)b * 3 * HW;
  float x0 = o[pix], x1 = o[(size_t)HW + pix], x2 = o[2 * (size_t)HW + pix];
  int   l  = lab[(size_t)b * HW + pix];
  float m  = fmaxf(x0, fmaxf(x1, x2));
  float lse = m + logf(expf(x0 - m) + expf(x1 - m) + expf(x2 - m));
  bool valid = (l != IGNORE_IDX);
  int  cl = valid ? (l < 0 ? 0 : (l > 2 ? 2 : l)) : 0;
  float xl = (cl == 0) ? x0 : ((cl == 1) ? x1 : x2);
  float nll = valid ? (lse - xl) : 0.0f;
  float cnt = valid ? 1.0f : 0.0f;
  for (int s = 16; s > 0; s >>= 1) { nll += __shfl_down(nll, s); cnt += __shfl_down(cnt, s); }
  __shared__ float sA[8], sB[8];
  int lane = threadIdx.x & 31, w = threadIdx.x >> 5;
  if (lane == 0) { sA[w] = nll; sB[w] = cnt; }
  __syncthreads();
  if (threadIdx.x == 0) {
    float a = 0, c = 0;
    for (int i = 0; i < 8; ++i) { a += sA[i]; c += sB[i]; }
    atomicAdd(&ce_sum[b], a);
    atomicAdd(&ce_cnt[b], c);
  }
}

// ---- Per (b,stream,pixel): strided softmax conf/argmax, norms, keys, masks ----
__global__ void prep_kernel(const float* __restrict__ out_m, const float* __restrict__ out_a,
                            const float* __restrict__ emb_m, const float* __restrict__ emb_a,
                            u32* __restrict__ keys, float* __restrict__ rnorms,
                            unsigned char* __restrict__ m2f,
                            u32* cnt1, u32* cnt2) {
  int s = blockIdx.y, b = blockIdx.z;
  int p = blockIdx.x * blockDim.x + threadIdx.x;     // < P (grid.x*256 == P)
  int i = p / Wew, j = p % Wew;
  const float* o = (s ? out_a : out_m) + (size_t)b * 3 * HW;
  size_t pix = (size_t)(i * 4) * Ww + (j * 4);
  float x0 = o[pix], x1 = o[(size_t)HW + pix], x2 = o[2 * (size_t)HW + pix];
  float m  = fmaxf(x0, fmaxf(x1, x2));
  int   seg  = (x0 == m) ? 0 : ((x1 == m) ? 1 : 2);   // first-max, like jnp.argmax
  float conf = 1.0f / (expf(x0 - m) + expf(x1 - m) + expf(x2 - m));

  const float* e = (s ? emb_a : emb_m) + (size_t)b * Dd * P + p;
  float sum = 0.0f, ss = 0.0f;
#pragma unroll 8
  for (int d = 0; d < Dd; ++d) { float v = e[(size_t)d * P]; sum += v; ss += v * v; }
  float rn  = 1.0f / sqrtf(ss);
  float key = sum * (1.0f / 64.0f);                  // channel-mean of raw embedding

  bool m1 = (seg == 1) && (conf > 0.8f);
  bool m2 = (seg == 2) && (conf > 0.6f);
  size_t bs = (size_t)(b * 2 + s);
  keys[bs * P + p]   = m1 ? f2sortable(key) : 0xFFFFFFFFu;
  rnorms[bs * P + p] = rn;
  m2f[bs * P + p]    = m2 ? 1 : 0;

  __shared__ u32 c1, c2;
  if (threadIdx.x == 0) { c1 = 0; c2 = 0; }
  __syncthreads();
  if (m1) atomicAdd(&c1, 1u);
  if (m2) atomicAdd(&c2, 1u);
  __syncthreads();
  if (threadIdx.x == 0) { atomicAdd(&cnt1[bs], c1); atomicAdd(&cnt2[bs], c2); }
}

// -------- Class-2: per-(b,s,d) masked sums of unit-vector components --------
__global__ void colsum2_kernel(const float* __restrict__ emb_m, const float* __restrict__ emb_a,
                               const float* __restrict__ rnorms, const unsigned char* __restrict__ m2f,
                               float* __restrict__ sum2) {
  int bs = blockIdx.z, d = blockIdx.y;
  int b = bs >> 1, s = bs & 1;
  int p = blockIdx.x * blockDim.x + threadIdx.x;
  const float* e = (s ? emb_a : emb_m) + (size_t)b * Dd * P + (size_t)d * P;
  float v = m2f[(size_t)bs * P + p] ? e[p] * rnorms[(size_t)bs * P + p] : 0.0f;
  for (int o = 16; o > 0; o >>= 1) v += __shfl_down(v, o);
  __shared__ float sA[8];
  int lane = threadIdx.x & 31, w = threadIdx.x >> 5;
  if (lane == 0) sA[w] = v;
  __syncthreads();
  if (threadIdx.x == 0) {
    float a = 0;
    for (int i = 0; i < 8; ++i) a += sA[i];
    atomicAdd(&sum2[(size_t)bs * Dd + d], a);
  }
}

// ---- Exact 400th-smallest key per (b,s): 4-pass radix select in LDS ----
__global__ void radix_kernel(const u32* __restrict__ keys, u32* thrT, u32* rneed) {
  int bs = blockIdx.x;
  const u32* k = keys + (size_t)bs * P;
  __shared__ u32 hist[256];
  __shared__ u32 s_prefix, s_remaining;
  if (threadIdx.x == 0) { s_prefix = 0; s_remaining = TOPK; }
  __syncthreads();
  for (int pass = 0; pass < 4; ++pass) {
    int shift = 24 - pass * 8;
    hist[threadIdx.x] = 0;
    __syncthreads();
    u32 pref   = s_prefix;
    u32 himask = (pass == 0) ? 0u : (0xFFFFFFFFu << (shift + 8));
    for (int p = threadIdx.x; p < P; p += 256) {
      u32 kv = k[p];
      if ((kv & himask) == pref) atomicAdd(&hist[(kv >> shift) & 255u], 1u);
    }
    __syncthreads();
    if (threadIdx.x == 0) {
      u32 rem = s_remaining, cum = 0; int sel = 255;
      for (int bin = 0; bin < 256; ++bin) {
        u32 c = hist[bin];
        if (cum + c >= rem) { sel = bin; break; }
        cum += c;
      }
      s_prefix    = pref | ((u32)sel << shift);
      s_remaining = rem - cum;
    }
    __syncthreads();
  }
  if (threadIdx.x == 0) { thrT[bs] = s_prefix; rneed[bs] = s_remaining; }
}

// ---- Deterministic select (ties by index order) + gather unit cols to f16 ----
__global__ void select_kernel(const u32* __restrict__ keys, const float* __restrict__ rnorms,
                              const float* __restrict__ emb_m, const float* __restrict__ emb_a,
                              const u32* __restrict__ thrT, const u32* __restrict__ rneed,
                              _Float16* __restrict__ S) {
  int bs = blockIdx.x, b = bs >> 1, s = bs & 1;
  u32 T = thrT[bs], r = rneed[bs];
  const u32*   k  = keys   + (size_t)bs * P;
  const float* rn = rnorms + (size_t)bs * P;
  const float* e  = (s ? emb_a : emb_m) + (size_t)b * Dd * P;
  _Float16* Sb = S + (size_t)bs * TOPK * Dd;

  __shared__ u32 wsum[8];
  __shared__ u32 base_eq, base_sel;
  if (threadIdx.x == 0) { base_eq = 0; base_sel = 0; }
  __syncthreads();
  int lane = threadIdx.x & 31, w = threadIdx.x >> 5;
  u32 lmask = (1u << lane) - 1u;

  for (int chunk = 0; chunk < P / 256; ++chunk) {
    int p = chunk * 256 + threadIdx.x;
    u32 kv = k[p];
    bool eq = (kv == T);
    unsigned em = (unsigned)(__ballot(eq) & 0xFFFFFFFFull);
    if (lane == 0) wsum[w] = __popc(em);
    __syncthreads();
    u32 eq_rank = base_eq + __popc(em & lmask);
    u32 eq_tot  = 0;
    for (int i = 0; i < 8; ++i) { if (i < w) eq_rank += wsum[i]; eq_tot += wsum[i]; }
    bool sel = (kv < T) || (eq && (eq_rank < r));
    __syncthreads();
    unsigned sm = (unsigned)(__ballot(sel) & 0xFFFFFFFFull);
    if (lane == 0) wsum[w] = __popc(sm);
    __syncthreads();
    u32 slot    = base_sel + __popc(sm & lmask);
    u32 sel_tot = 0;
    for (int i = 0; i < 8; ++i) { if (i < w) slot += wsum[i]; sel_tot += wsum[i]; }
    if (sel && slot < (u32)TOPK) {
      float rv = rn[p];
#pragma unroll 8
      for (int d = 0; d < Dd; ++d)
        Sb[(size_t)slot * Dd + d] = (_Float16)(e[(size_t)d * P + p] * rv);
    }
    __syncthreads();
    if (threadIdx.x == 0) { base_eq += eq_tot; base_sel += sel_tot; }
    __syncthreads();
  }
}

// ---- WMMA pairwise cosine sums: per sample, sum over all S1^T * S1a ----
// Column storage: 64 contiguous f16 per selected pixel (128B, 16B aligned).
// A 16x32 f16 fragment (ISA 7.12.2): lanes 0-15 -> M=lane, K chunks {0..7,16..23};
// lanes 16-31 -> M=lane-16, K chunks {8..15,24..31}. B mirrors with N<->M.
__device__ __forceinline__ half16 load_frag(const _Float16* col, int kbase, int grp) {
  const half8 lo = *(const half8*)(col + kbase + grp * 8);
  const half8 hi = *(const half8*)(col + kbase + 16 + grp * 8);
  half16 f;
#pragma unroll
  for (int t = 0; t < 8; ++t) { f[t] = lo[t]; f[8 + t] = hi[t]; }
  return f;
}

__global__ void pair_kernel(const _Float16* __restrict__ S, float* __restrict__ pairsum) {
  int wid  = (int)((blockIdx.x * blockDim.x + threadIdx.x) >> 5); // 0..9999 exactly
  int lane = threadIdx.x & 31;
  int b  = wid / (NTILE * NTILE);
  int t  = wid % (NTILE * NTILE);
  int ti = t / NTILE, tj = t % NTILE;
  int r = lane & 15, grp = lane >> 4;

  const _Float16* Acol = S + ((size_t)(b * 2 + 0) * TOPK + ti * 16 + r) * Dd;
  const _Float16* Bcol = S + ((size_t)(b * 2 + 1) * TOPK + tj * 16 + r) * Dd;

  half16 a0 = load_frag(Acol, 0, grp),  b0 = load_frag(Bcol, 0, grp);
  half16 a1 = load_frag(Acol, 32, grp), b1 = load_frag(Bcol, 32, grp);

  float8 c = {};
  c = __builtin_amdgcn_wmma_f32_16x16x32_f16(false, a0, false, b0, (short)0, c, false, false);
  c = __builtin_amdgcn_wmma_f32_16x16x32_f16(false, a1, false, b1, (short)0, c, false, false);

  float sum = c[0] + c[1] + c[2] + c[3] + c[4] + c[5] + c[6] + c[7];
  for (int o = 16; o > 0; o >>= 1) sum += __shfl_down(sum, o);
  if (lane == 0) atomicAdd(&pairsum[b], sum);
}

// ---------------------------- Final combine ----------------------------
__global__ void finalize_kernel(const float* ce_sum, const float* ce_cnt,
                                const u32* cnt1, const u32* cnt2,
                                const float* sum2, const float* pairsum,
                                float* __restrict__ out) {
  float ce = 0.0f;
  for (int b = 0; b < Bn; ++b) ce += ce_sum[b] / fmaxf(ce_cnt[b], 1.0f);
  ce *= (1.0f / (float)Bn);

  float tot = 0.0f; int cnt = 0;
  for (int b = 0; b < Bn; ++b) {
    bool v1 = (cnt1[b * 2] > (u32)TOPK) && (cnt1[b * 2 + 1] > (u32)TOPK);
    if (v1) { tot += 1.0f - pairsum[b] * (1.0f / ((float)TOPK * (float)TOPK)); ++cnt; }
    u32 c0 = cnt2[b * 2], c1 = cnt2[b * 2 + 1];
    float i0 = 1.0f / (float)(c0 > 0 ? c0 : 1);
    float i1 = 1.0f / (float)(c1 > 0 ? c1 : 1);
    float dot = 0.0f;
    for (int d = 0; d < Dd; ++d)
      dot += (sum2[(size_t)(b * 2) * Dd + d] * i0) * (sum2[(size_t)(b * 2 + 1) * Dd + d] * i1);
    if (c0 > 0 && c1 > 0) { tot += 1.0f - dot; ++cnt; }
  }
  float lm = tot / fmaxf((float)cnt, 1.0f);
  out[0] = ce + 2.0f * lm;
  out[1] = ce;
  out[2] = lm;
}

extern "C" void kernel_launch(void* const* d_in, const int* in_sizes, int n_in,
                              void* d_out, int out_size, void* d_ws, size_t ws_size,
                              hipStream_t stream) {
  const float* outputs   = (const float*)d_in[0];
  const float* emb       = (const float*)d_in[1];
  const int*   labels    = (const int*)d_in[2];
  const float* outputs_a = (const float*)d_in[3];
  const float* emb_a     = (const float*)d_in[4];
  // d_in[5] (class_labels_aug) does not enter the loss.
  float* out = (float*)d_out;

  char* w = (char*)d_ws;
  float* ce_sum  = (float*)(w + OFF_CESUM);
  float* ce_cnt  = (float*)(w + OFF_CECNT);
  u32*   cnt1    = (u32*)  (w + OFF_CNT1);
  u32*   cnt2    = (u32*)  (w + OFF_CNT2);
  float* pairsum = (float*)(w + OFF_PAIR);
  float* sum2    = (float*)(w + OFF_SUM2);
  u32*   thrT    = (u32*)  (w + OFF_THR);
  u32*   rneed   = (u32*)  (w + OFF_RNEED);
  u32*   keys    = (u32*)  (w + OFF_KEYS);
  float* rnorms  = (float*)(w + OFF_RNORM);
  unsigned char* m2f = (unsigned char*)(w + OFF_M2);
  _Float16* Smat = (_Float16*)(w + OFF_SMAT);

  hipMemsetAsync(w, 0, ACC_BYTES, stream);  // zero accumulators every call (graph-safe)

  ce_kernel<<<dim3(HW / 256, Bn), 256, 0, stream>>>(outputs, labels, ce_sum, ce_cnt);
  prep_kernel<<<dim3(P / 256, 2, Bn), 256, 0, stream>>>(outputs, outputs_a, emb, emb_a,
                                                        keys, rnorms, m2f, cnt1, cnt2);
  colsum2_kernel<<<dim3(P / 256, Dd, 2 * Bn), 256, 0, stream>>>(emb, emb_a, rnorms, m2f, sum2);
  radix_kernel<<<2 * Bn, 256, 0, stream>>>(keys, thrT, rneed);
  select_kernel<<<2 * Bn, 256, 0, stream>>>(keys, rnorms, emb, emb_a, thrT, rneed, Smat);
  pair_kernel<<<(Bn * NTILE * NTILE) / 8, 256, 0, stream>>>(Smat, pairsum);
  finalize_kernel<<<1, 1, 0, stream>>>(ce_sum, ce_cnt, cnt1, cnt2, sum2, pairsum, out);
}
